// SelfAttention_35467839930992
// MI455X (gfx1250) — matrix operand order repaired and checked by reading, not compiled, not run on hardware
//
#include <hip/hip_runtime.h>

// ---------------------------------------------------------------------------
// Self-attention for MI455X (gfx1250): bf16 WMMA GEMMs + flash-attention core.
// CDNA5 features: v_wmma_f32_16x16x32_bf16, global_load_async_to_lds_b128
// (ASYNCcnt-tracked LDS staging, double-buffered GEMM), global_prefetch_b8.
// ---------------------------------------------------------------------------

typedef __attribute__((ext_vector_type(16))) __bf16 v16bf;
typedef __attribute__((ext_vector_type(8)))  float  v8f;

union Frag16 { v16bf v; unsigned u[8]; };

static __device__ __forceinline__ unsigned short f2bf(float f) {
    unsigned u = __float_as_uint(f);
    unsigned r = (u + 0x7FFFu + ((u >> 16) & 1u)) >> 16;
    return (unsigned short)r;
}

static __device__ __forceinline__ v8f vzero8() {
    v8f z = {0.f, 0.f, 0.f, 0.f, 0.f, 0.f, 0.f, 0.f};
    return z;
}

// Async copy 16B global -> LDS (ASYNCcnt). lds_off = low 32 bits of the
// generic shared-space address (== LDS byte offset per ISA aperture mapping).
static __device__ __forceinline__ void async_ld_b128(void* lds_ptr, const void* gptr) {
    unsigned lds_off = (unsigned)(uintptr_t)lds_ptr;
    asm volatile("global_load_async_to_lds_b128 %0, %1, off"
                 :: "v"(lds_off), "v"(gptr) : "memory");
}
static __device__ __forceinline__ void wait_async0() {
    asm volatile("s_wait_asynccnt 0" ::: "memory");
}

// A-matrix 16x32 bf16 K-offsets per VGPR p (lanes 0-15 add 0, lanes 16-31 add 8)
__device__ __constant__ int kbA[8] = {0, 2, 4, 6, 16, 18, 20, 22};

// ---------------------------------------------------------------------------
// fp32 -> bf16 bulk convert
// ---------------------------------------------------------------------------
__global__ void cvt_f32_bf16(const float* __restrict__ in,
                             unsigned short* __restrict__ out, int n) {
    int i = blockIdx.x * blockDim.x + threadIdx.x;
    int stride = gridDim.x * blockDim.x;
    for (; i < n; i += stride) out[i] = f2bf(in[i]);
}

// Weight [K=1024][N=1024] fp32 -> bf16 with k-pair interleave:
// out[(k>>1)*2048 + n*2 + (k&1)]  so B-fragment pairs (k,k+1) are contiguous.
__global__ void cvt_weight_swz(const float* __restrict__ w,
                               unsigned short* __restrict__ out) {
    int i = blockIdx.x * blockDim.x + threadIdx.x;
    int stride = gridDim.x * blockDim.x;
    for (; i < 1024 * 1024; i += stride) {
        int k = i >> 10, n = i & 1023;
        out[(size_t)(k >> 1) * 2048 + n * 2 + (k & 1)] = f2bf(w[i]);
    }
}

// ---------------------------------------------------------------------------
// Tiled bf16 WMMA GEMM: C[M x 1024] = A[M x 1024] * Bswz + bias, then scale.
// Block tile 128x128, 256 threads = 8 waves (4 M-waves x 2 N-waves),
// wave tile 32x64 -> 2x4 accumulators. Double-buffered LDS staged with
// global_load_async_to_lds_b128 so copies overlap the WMMA stream.
// MODE 0: write bf16 scattered to per-head [B,H,S,64] layout (Q/K/V)
// MODE 1: write fp32 row-major (final output projection)
// ---------------------------------------------------------------------------
#define ASTR 40   // A tile LDS stride (elems): 80B rows, 16B aligned, conflict-free

template <int MODE>
__global__ __launch_bounds__(256)
void gemm_bf16_wmma(const unsigned short* __restrict__ A,
                    const unsigned short* __restrict__ Bswz,
                    const float* __restrict__ bias,
                    unsigned short* __restrict__ out16,
                    float* __restrict__ out32,
                    float scale) {
    __shared__ unsigned short As[2][128 * ASTR];
    __shared__ unsigned short Bs[2][16 * 256];

    const int tid = threadIdx.x;
    const int w = tid >> 5, lane = tid & 31;
    const int hl = lane >> 4, l16 = lane & 15;
    const int wm = w & 3, wn = w >> 2;
    const int m0 = blockIdx.y * 128, n0 = blockIdx.x * 128;

    // Per-thread staging coordinates (2 x 16B chunks for A, 2 for B)
    const int arow0 = tid >> 2,          acol0 = (tid & 3) * 8;
    const int arow1 = (tid + 256) >> 2,  acol1 = acol0;
    const int brow0 = tid >> 5,          bcol0 = (tid & 31) * 8;
    const int brow1 = brow0 + 8,         bcol1 = bcol0;

    auto stage = [&](int kt, int buf) {
        const int k0 = kt * 32;
        async_ld_b128(&As[buf][arow0 * ASTR + acol0],
                      &A[(size_t)(m0 + arow0) * 1024 + k0 + acol0]);
        async_ld_b128(&As[buf][arow1 * ASTR + acol1],
                      &A[(size_t)(m0 + arow1) * 1024 + k0 + acol1]);
        async_ld_b128(&Bs[buf][brow0 * 256 + bcol0],
                      &Bswz[(size_t)((k0 >> 1) + brow0) * 2048 + n0 * 2 + bcol0]);
        async_ld_b128(&Bs[buf][brow1 * 256 + bcol1],
                      &Bswz[(size_t)((k0 >> 1) + brow1) * 2048 + n0 * 2 + bcol1]);
    };

    v8f acc[2][4];
#pragma unroll
    for (int mi = 0; mi < 2; ++mi)
#pragma unroll
        for (int ni = 0; ni < 4; ++ni) acc[mi][ni] = vzero8();

    stage(0, 0);
    wait_async0();
    __syncthreads();

    for (int kt = 0; kt < 32; ++kt) {
        const int buf = kt & 1;
        if (kt < 31) stage(kt + 1, buf ^ 1);  // overlap copy with compute

        Frag16 a[2], b[4];
#pragma unroll
        for (int mi = 0; mi < 2; ++mi) {
            int row = wm * 32 + mi * 16 + l16;
#pragma unroll
            for (int p = 0; p < 8; ++p) {
                int k = kbA[p] + hl * 8;
                a[mi].u[p] = *(const unsigned*)&As[buf][row * ASTR + k];
            }
        }
#pragma unroll
        for (int ni = 0; ni < 4; ++ni) {
            int n = wn * 64 + ni * 16 + l16;
#pragma unroll
            for (int p = 0; p < 8; ++p) {
                int k = kbA[p] + hl * 8;   // B layout: same k order as A (kbA + 8 per lane half)
                b[ni].u[p] = *(const unsigned*)&Bs[buf][(k >> 1) * 256 + n * 2];
            }
        }
#pragma unroll
        for (int mi = 0; mi < 2; ++mi)
#pragma unroll
            for (int ni = 0; ni < 4; ++ni)
                acc[mi][ni] = __builtin_amdgcn_wmma_f32_16x16x32_bf16(
                    false, a[mi].v, false, b[ni].v, (short)0, acc[mi][ni],
                    false, false);
        wait_async0();
        __syncthreads();
    }

    // Epilogue (branch-free per template instantiation)
#pragma unroll
    for (int mi = 0; mi < 2; ++mi)
#pragma unroll
        for (int ni = 0; ni < 4; ++ni) {
            int nn = n0 + wn * 64 + ni * 16 + l16;
            float bv = bias[nn];
#pragma unroll
            for (int r = 0; r < 8; ++r) {
                int mm = m0 + wm * 32 + mi * 16 + hl * 8 + r;
                float v = (acc[mi][ni][r] + bv) * scale;
                if (MODE == 0) {
                    int bb = mm >> 11, ss = mm & 2047;
                    int hh = nn >> 6, dd = nn & 63;
                    out16[((size_t)(bb * 16 + hh) * 2048 + ss) * 64 + dd] = f2bf(v);
                } else {
                    out32[(size_t)mm * 1024 + nn] = v;
                }
            }
        }
}

// ---------------------------------------------------------------------------
// Flash attention: grid (S/64, B*H), 128 threads = 4 waves, each wave = 16 q rows.
// K tile staged via async-to-LDS; V transposed through registers; per 64-wide
// KV tile: 8 WMMAs for Q*K^T, online softmax (butterfly shuffles in 16-lane
// halves), P -> LDS (C-layout -> A-layout), 8 WMMAs for P*V.
// ---------------------------------------------------------------------------
#define KSTR 72   // Kt row stride (elems): 144B rows (16B-aligned), conflict-free
#define VSTR 66   // Vt (transposed) row stride
#define PSTR 66   // per-wave P tile stride

__global__ __launch_bounds__(128)
void flash_attn_wmma(const unsigned short* __restrict__ Q,
                     const unsigned short* __restrict__ K,
                     const unsigned short* __restrict__ V,
                     unsigned short* __restrict__ Oc) {
    __shared__ unsigned short Kt[64 * KSTR];
    __shared__ unsigned short Vt[64 * VSTR];
    __shared__ unsigned short Pt[4][16 * PSTR];

    const int tid = threadIdx.x;
    const int w = tid >> 5, lane = tid & 31;
    const int hl = lane >> 4, l16 = lane & 15;
    const int bh = blockIdx.y;
    const int q0 = blockIdx.x * 64 + w * 16;

    const unsigned short* Qh = Q + (size_t)bh * 2048 * 64;
    const unsigned short* Kh = K + (size_t)bh * 2048 * 64;
    const unsigned short* Vh = V + (size_t)bh * 2048 * 64;

    // Q fragments (A layout), once per wave
    Frag16 qf[2];
#pragma unroll
    for (int ks = 0; ks < 2; ++ks)
#pragma unroll
        for (int p = 0; p < 8; ++p) {
            int k = ks * 32 + kbA[p] + hl * 8;
            qf[ks].u[p] = *(const unsigned*)&Qh[(size_t)(q0 + l16) * 64 + k];
        }

    float mrow[8], lrow[8];
    v8f oacc[4];
#pragma unroll
    for (int r = 0; r < 8; ++r) { mrow[r] = -1e30f; lrow[r] = 0.f; }
#pragma unroll
    for (int ni = 0; ni < 4; ++ni) oacc[ni] = vzero8();

    for (int s0 = 0; s0 < 2048; s0 += 64) {
        // Stage K tile via async copy; V tile transposed through registers.
#pragma unroll
        for (int it = 0; it < 4; ++it) {
            int c = tid + it * 128;
            int row = c >> 3, d0 = (c & 7) * 8;
            async_ld_b128(&Kt[row * KSTR + d0], &Kh[(size_t)(s0 + row) * 64 + d0]);
            union { uint4 q; unsigned short s[8]; } t;
            t.q = *(const uint4*)&Vh[(size_t)(s0 + row) * 64 + d0];
#pragma unroll
            for (int e = 0; e < 8; ++e) Vt[(d0 + e) * VSTR + row] = t.s[e];
        }
        // Prefetch next tiles into cache while this one computes.
        if (s0 + 64 < 2048) {
            __builtin_prefetch(&Kh[(size_t)(s0 + 64 + (tid >> 1)) * 64], 0, 0);
            __builtin_prefetch(&Vh[(size_t)(s0 + 64 + (tid >> 1)) * 64], 0, 0);
        }
        wait_async0();
        __syncthreads();

        // Scores: S = Q * K^T  (Q pre-scaled by 1/sqrt(64))
        v8f sc[4];
#pragma unroll
        for (int ni = 0; ni < 4; ++ni) sc[ni] = vzero8();
#pragma unroll
        for (int ks = 0; ks < 2; ++ks)
#pragma unroll
            for (int ni = 0; ni < 4; ++ni) {
                Frag16 kf;
                int n = ni * 16 + l16;  // kv column
#pragma unroll
                for (int p = 0; p < 8; ++p) {
                    int k = ks * 32 + kbA[p] + hl * 8;  // d index; pairs contiguous
                    kf.u[p] = *(const unsigned*)&Kt[n * KSTR + k];
                }
                sc[ni] = __builtin_amdgcn_wmma_f32_16x16x32_bf16(
                    false, qf[ks].v, false, kf.v, (short)0, sc[ni], false, false);
            }

        // Online softmax: rows live in VGPR index r, spread over a 16-lane half
        float mnew[8], alpha[8];
#pragma unroll
        for (int r = 0; r < 8; ++r) {
            float v = fmaxf(fmaxf(sc[0][r], sc[1][r]), fmaxf(sc[2][r], sc[3][r]));
#pragma unroll
            for (int off = 1; off < 16; off <<= 1)
                v = fmaxf(v, __shfl_xor(v, off, 32));
            mnew[r] = fmaxf(mrow[r], v);
            alpha[r] = __expf(mrow[r] - mnew[r]);
            mrow[r] = mnew[r];
        }
        unsigned short* Pw = &Pt[w][0];
#pragma unroll
        for (int ni = 0; ni < 4; ++ni)
#pragma unroll
            for (int r = 0; r < 8; ++r) {
                float p = __expf(sc[ni][r] - mnew[r]);
                sc[ni][r] = p;
                Pw[(hl * 8 + r) * PSTR + ni * 16 + l16] = f2bf(p);
            }
#pragma unroll
        for (int r = 0; r < 8; ++r) {
            float s = sc[0][r] + sc[1][r] + sc[2][r] + sc[3][r];
#pragma unroll
            for (int off = 1; off < 16; off <<= 1)
                s += __shfl_xor(s, off, 32);
            lrow[r] = lrow[r] * alpha[r] + s;
        }
        __syncthreads();

        // Rescale accumulators, then O += P * V
#pragma unroll
        for (int ni = 0; ni < 4; ++ni)
#pragma unroll
            for (int r = 0; r < 8; ++r) oacc[ni][r] *= alpha[r];

        Frag16 pf[2];
#pragma unroll
        for (int ks = 0; ks < 2; ++ks)
#pragma unroll
            for (int p = 0; p < 8; ++p) {
                int k = ks * 32 + kbA[p] + hl * 8;
                pf[ks].u[p] = *(const unsigned*)&Pw[l16 * PSTR + k];
            }
#pragma unroll
        for (int ks = 0; ks < 2; ++ks)
#pragma unroll
            for (int ni = 0; ni < 4; ++ni) {
                Frag16 vf;
                int d = ni * 16 + l16;
#pragma unroll
                for (int p = 0; p < 8; ++p) {
                    int k = ks * 32 + kbA[p] + hl * 8;  // kv; pairs contiguous in Vt
                    vf.u[p] = *(const unsigned*)&Vt[d * VSTR + k];
                }
                oacc[ni] = __builtin_amdgcn_wmma_f32_16x16x32_bf16(
                    false, pf[ks].v, false, vf.v, (short)0, oacc[ni], false, false);
            }
        __syncthreads();
    }

    // Write context to [B, S, H*64] bf16 for the output projection GEMM
    const int b = bh >> 4, h = bh & 15;
#pragma unroll
    for (int ni = 0; ni < 4; ++ni)
#pragma unroll
        for (int r = 0; r < 8; ++r) {
            int s = q0 + hl * 8 + r;
            float v = oacc[ni][r] / lrow[r];
            Oc[((size_t)(b * 2048 + s)) * 1024 + h * 64 + ni * 16 + l16] = f2bf(v);
        }
}

// ---------------------------------------------------------------------------
// Host launcher
// ---------------------------------------------------------------------------
extern "C" void kernel_launch(void* const* d_in, const int* in_sizes, int n_in,
                              void* d_out, int out_size, void* d_ws, size_t ws_size,
                              hipStream_t stream) {
    const float* x   = (const float*)d_in[0];
    const float* q_w = (const float*)d_in[1];
    const float* q_b = (const float*)d_in[2];
    const float* k_w = (const float*)d_in[3];
    const float* k_b = (const float*)d_in[4];
    const float* v_w = (const float*)d_in[5];
    const float* v_b = (const float*)d_in[6];
    const float* o_w = (const float*)d_in[7];
    const float* o_b = (const float*)d_in[8];

    const size_t SZ_X  = (size_t)8192 * 1024 * 2;    // 16 MB bf16
    const size_t SZ_W  = (size_t)1024 * 1024 * 2;    // 2 MB bf16
    const size_t SZ_H  = (size_t)64 * 2048 * 64 * 2; // 16 MB bf16 per Q/K/V

    char* p = (char*)d_ws;
    unsigned short* xb    = (unsigned short*)p; p += SZ_X;
    unsigned short* wq    = (unsigned short*)p; p += SZ_W;
    unsigned short* wk    = (unsigned short*)p; p += SZ_W;
    unsigned short* wv    = (unsigned short*)p; p += SZ_W;
    unsigned short* wo    = (unsigned short*)p; p += SZ_W;
    unsigned short* Qb    = (unsigned short*)p; p += SZ_H;
    unsigned short* Kb    = (unsigned short*)p; p += SZ_H;
    unsigned short* Vb    = (unsigned short*)p; p += SZ_H;
    unsigned short* attnb = (unsigned short*)p; p += SZ_X;

    if ((size_t)(p - (char*)d_ws) > ws_size) return;  // scratch too small

    // Precision conversion / weight swizzle
    cvt_f32_bf16<<<8192, 256, 0, stream>>>(x, xb, 8192 * 1024);
    cvt_weight_swz<<<2048, 256, 0, stream>>>(q_w, wq);
    cvt_weight_swz<<<2048, 256, 0, stream>>>(k_w, wk);
    cvt_weight_swz<<<2048, 256, 0, stream>>>(v_w, wv);
    cvt_weight_swz<<<2048, 256, 0, stream>>>(o_w, wo);

    // Q/K/V projections (Q pre-scaled by 1/sqrt(head_dim) = 0.125)
    dim3 ggrid(8, 64);  // N/128, M/128
    gemm_bf16_wmma<0><<<ggrid, 256, 0, stream>>>(xb, wq, q_b, Qb, nullptr, 0.125f);
    gemm_bf16_wmma<0><<<ggrid, 256, 0, stream>>>(xb, wk, k_b, Kb, nullptr, 1.0f);
    gemm_bf16_wmma<0><<<ggrid, 256, 0, stream>>>(xb, wv, v_b, Vb, nullptr, 1.0f);

    // Flash attention over 64 (B*H) head-batches, 32 q-tiles of 64 rows
    flash_attn_wmma<<<dim3(32, 64), 128, 0, stream>>>(Qb, Kb, Vb, attnb);

    // Output projection -> fp32 d_out
    gemm_bf16_wmma<1><<<ggrid, 256, 0, stream>>>(attnb, wo, o_b, nullptr,
                                                 (float*)d_out, 1.0f);
}
